// MultiHeadAttention_26268019982958
// MI455X (gfx1250) — matrix operand order repaired and checked by reading, not compiled, not run on hardware
//
#include <hip/hip_runtime.h>
#include <hip/hip_bf16.h>

// ---------------------------------------------------------------------------
// MI455X (gfx1250) multi-head attention: bf16 WMMA everywhere.
//   x:[4,2048,2048] f32, Wq/Wk/Wv/Wo:[2048,2048] f32  ->  out:[4,2048,2048] f32
// ---------------------------------------------------------------------------

typedef unsigned short u16;
typedef unsigned int   u32;
typedef unsigned long long u64;
typedef __attribute__((ext_vector_type(16))) __bf16 v16bf;
typedef __attribute__((ext_vector_type(8)))  __bf16 v8bf;
typedef __attribute__((ext_vector_type(8)))  float  v8f;
typedef __attribute__((ext_vector_type(4)))  int    v4i;

#define HIDDEN   2048
#define HEADS    16
#define HEAD_DIM 128
#define SEQ      2048
#define BATCH    4
#define MROWS    (BATCH * SEQ)   // 8192

// ---- async global->LDS (CDNA5 GLOBAL_LOAD_ASYNC_TO_LDS_B128, ASYNCcnt) ----
#if defined(__has_builtin)
#if __has_builtin(__builtin_amdgcn_global_load_async_to_lds_b128)
#define HAVE_ASYNC_LDS 1
#endif
#endif
#ifndef HAVE_ASYNC_LDS
#define HAVE_ASYNC_LDS 0
#endif

#if HAVE_ASYNC_LDS
typedef __attribute__((address_space(3))) v4i lds_v4i;
__device__ __forceinline__ void async_copy_b128(void* l, const void* g) {
  // prototype (from hipcc diagnostic): param0 = generic v4i* (global address),
  // param1 = LDS v4i*, then imm offset, imm cpol.
  __builtin_amdgcn_global_load_async_to_lds_b128(
      (v4i*)g, (lds_v4i*)(u32)(u64)l, 0, 0);
}
#if __has_builtin(__builtin_amdgcn_s_wait_asynccnt)
#define ASYNC_WAIT() __builtin_amdgcn_s_wait_asynccnt(0)
#else
#define ASYNC_WAIT() asm volatile("s_wait_asynccnt 0x0" ::: "memory")
#endif
#else
__device__ __forceinline__ void async_copy_b128(void* l, const void* g) {
  *(uint4*)l = *(const uint4*)g;
}
#define ASYNC_WAIT() ((void)0)
#endif

// ---- scalar bf16 <-> f32 helpers (round-to-nearest-even) -------------------
__device__ __forceinline__ u16 f2bf(float f) {
  u32 u = __builtin_bit_cast(u32, f);
  u += 0x7fffu + ((u >> 16) & 1u);
  return (u16)(u >> 16);
}
__device__ __forceinline__ float bf2f(u16 h) {
  u32 u = ((u32)h) << 16;
  return __builtin_bit_cast(float, u);
}
__device__ __forceinline__ v8f vzero8() {
  v8f z;
#pragma unroll
  for (int i = 0; i < 8; ++i) z[i] = 0.0f;
  return z;
}

// ---- WMMA wrapper -----------------------------------------------------------
__device__ __forceinline__ v8f wmma_bf16(v16bf a, v16bf b, v8f c) {
  return __builtin_amdgcn_wmma_f32_16x16x32_bf16(
      false, a, false, b, (short)0, c, false, false);
}

// A fragment (16x32, 16-bit) from row-major LDS tile, stride in elements.
__device__ __forceinline__ v16bf load_a_frag(const u16* lds, int row_base,
                                             int stride, int k0) {
  const int lane = threadIdx.x & 31;
  const int m = lane & 15, h = lane >> 4;
  const u16* p = lds + (size_t)(row_base + m) * stride + k0;
  v8bf lo = *(const v8bf*)(p + h * 8);
  v8bf hi = *(const v8bf*)(p + 16 + h * 8);
  return __builtin_shufflevector(lo, hi, 0, 1, 2, 3, 4, 5, 6, 7,
                                 8, 9, 10, 11, 12, 13, 14, 15);
}

// B fragment (32x16, 16-bit) from TRANSPOSED LDS tile Bt[n][k].
__device__ __forceinline__ v16bf load_b_frag(const u16* ldsBt, int col_base,
                                             int stride, int k0) {
  const int lane = threadIdx.x & 31;
  const int n = lane & 15, h = lane >> 4;
  const u16* p = ldsBt + (size_t)(col_base + n) * stride + k0;
  v8bf lo = *(const v8bf*)(p + h * 8);
  v8bf hi = *(const v8bf*)(p + 16 + h * 8);
  return __builtin_shufflevector(lo, hi, 0, 1, 2, 3, 4, 5, 6, 7,
                                 8, 9, 10, 11, 12, 13, 14, 15);
}

// ---------------------------------------------------------------------------
// Kernel 1: f32 -> bf16 convert (grid-stride)
// ---------------------------------------------------------------------------
__global__ void cvt_f32_bf16_kernel(const float* __restrict__ in,
                                    u16* __restrict__ out, size_t n) {
  size_t i = (size_t)blockIdx.x * blockDim.x + threadIdx.x;
  const size_t stride = (size_t)gridDim.x * blockDim.x;
  for (; i < n; i += stride) out[i] = f2bf(in[i]);
}

// ---------------------------------------------------------------------------
// Kernel 2: bf16 GEMM  C[M,N] = A[M,K] @ B[K,N]
//   256x128 block tile, 8 waves (4Mx2N), 64x64 per wave, KB=64,
//   double-buffered LDS with async global->LDS staging for A.
// ---------------------------------------------------------------------------
__global__ __launch_bounds__(256) void gemm_bf16_kernel(
    const u16* __restrict__ A, const u16* __restrict__ Bm,
    u16* __restrict__ Cb, float* __restrict__ Cf,
    int M, int N, int K) {
  __shared__ u16 sA[2][256 * 64];    // 2 x 32 KB
  __shared__ u16 sBt[2][128 * 64];   // 2 x 16 KB  (transposed [n][k])

  const int t = threadIdx.x;
  const int wid = t >> 5;
  const int wm = wid & 3;            // 0..3 -> 64-row strip
  const int wn = wid >> 2;           // 0..1 -> 64-col strip
  const int bm = blockIdx.y * 256;
  const int bn = blockIdx.x * 128;

  v8f acc[4][4];
#pragma unroll
  for (int i = 0; i < 4; ++i)
#pragma unroll
    for (int j = 0; j < 4; ++j) acc[i][j] = vzero8();

  // --- staging helpers ------------------------------------------------------
  auto stageA = [&](int buf, int k0) {      // 256x64 = 2048 x 16B chunks
#pragma unroll
    for (int c = 0; c < 8; ++c) {
      int idx = t + c * 256;
      int r = idx >> 3, cg = (idx & 7) * 8;
      async_copy_b128(&sA[buf][r * 64 + cg],
                      &A[(size_t)(bm + r) * K + k0 + cg]);
    }
  };
  uint4 breg[4];
  auto loadB = [&](int k0) {                // 64 k-rows x 16 n-chunks
#pragma unroll
    for (int c = 0; c < 4; ++c) {
      int idx = t + c * 256;
      int kk = idx >> 4, ng = (idx & 15) * 8;
      breg[c] = *(const uint4*)&Bm[(size_t)(k0 + kk) * N + bn + ng];
    }
  };
  auto storeB = [&](int buf) {              // scatter-transpose into Bt[n][k]
#pragma unroll
    for (int c = 0; c < 4; ++c) {
      int idx = t + c * 256;
      int kk = idx >> 4, ng = (idx & 15) * 8;
      const u16* e = (const u16*)&breg[c];
#pragma unroll
      for (int j = 0; j < 8; ++j) sBt[buf][(ng + j) * 64 + kk] = e[j];
    }
  };

  // --- prologue: stage 0 ----------------------------------------------------
  stageA(0, 0);
  loadB(0);
  storeB(0);

  const int nstage = K / 64;
  for (int s = 0; s < nstage; ++s) {
    const int cur = s & 1, nxt = cur ^ 1;
    ASYNC_WAIT();          // my async A-copies for 'cur' complete
    __syncthreads();       // everyone staged 'cur'; everyone done with 'nxt'

    if (s + 1 < nstage) {  // overlap next stage's global traffic with WMMA
      stageA(nxt, (s + 1) * 64);
      loadB((s + 1) * 64);
    }

#pragma unroll
    for (int ks = 0; ks < 64; ks += 32) {
      v16bf a[4];
#pragma unroll
      for (int i = 0; i < 4; ++i)
        a[i] = load_a_frag(sA[cur], wm * 64 + i * 16, 64, ks);
#pragma unroll
      for (int j = 0; j < 4; ++j) {
        v16bf bj = load_b_frag(sBt[cur], wn * 64 + j * 16, 64, ks);
#pragma unroll
        for (int i = 0; i < 4; ++i) acc[i][j] = wmma_bf16(a[i], bj, acc[i][j]);
      }
    }

    if (s + 1 < nstage) storeB(nxt);  // ds-stores into 'nxt' (safe: barrier passed)
  }

  // --- store: C layout: lane&15 = col, (lane>>4)*8 + r = row ----------------
  const int lane = t & 31;
  const int cN = lane & 15;
  const int rb = (lane >> 4) * 8;
#pragma unroll
  for (int i = 0; i < 4; ++i)
#pragma unroll
    for (int j = 0; j < 4; ++j)
#pragma unroll
      for (int r = 0; r < 8; ++r) {
        int row = bm + wm * 64 + i * 16 + rb + r;
        int col = bn + wn * 64 + j * 16 + cN;
        float v = acc[i][j][r];
        if (Cb) Cb[(size_t)row * N + col] = f2bf(v);
        else    Cf[(size_t)row * N + col] = v;
      }
}

// ---------------------------------------------------------------------------
// Kernel 3: RoPE in place on bf16 Q and K.
// ---------------------------------------------------------------------------
__global__ __launch_bounds__(256) void rope_kernel(u16* __restrict__ Q,
                                                   u16* __restrict__ Kb) {
  size_t tid = (size_t)blockIdx.x * blockDim.x + threadIdx.x;
  const int i = (int)(tid & 63);
  size_t rest = tid >> 6;
  const int h = (int)(rest % HEADS); rest /= HEADS;
  const int l = (int)(rest % SEQ);
  const int b = (int)(rest / SEQ);
  const size_t base = ((size_t)(b * SEQ + l)) * HIDDEN + (size_t)h * HEAD_DIM;

  const float inv_freq = __powf(10000.0f, -(float)i / 64.0f);
  float s, c;
  __sincosf((float)l * inv_freq, &s, &c);

  float q1 = bf2f(Q[base + i]), q2 = bf2f(Q[base + 64 + i]);
  Q[base + i]      = f2bf(q1 * c - q2 * s);
  Q[base + 64 + i] = f2bf(q2 * c + q1 * s);

  float k1 = bf2f(Kb[base + i]), k2 = bf2f(Kb[base + 64 + i]);
  Kb[base + i]      = f2bf(k1 * c - k2 * s);
  Kb[base + 64 + i] = f2bf(k2 * c + k1 * s);
}

// ---------------------------------------------------------------------------
// Kernel 4: causal flash attention, one workgroup per (q-128-block, b*H+h).
// ---------------------------------------------------------------------------
__global__ __launch_bounds__(256) void attn_kernel(
    const u16* __restrict__ Q, const u16* __restrict__ Kg,
    const u16* __restrict__ V, u16* __restrict__ Og) {
  extern __shared__ char smem[];
  u16*   Qs   = (u16*)smem;               // [128][128] bf16 (32 KB)
  u16*   Ks   = Qs + 128 * 128;           // [kv][hd]        (32 KB) == Bt for QK^T
  u16*   Vt   = Ks + 128 * 128;           // [hd][kv]        (32 KB) == Bt for P*V
  u16*   Ps   = Vt + 128 * 128;           // [q][kv]         (32 KB)
  float* Sf   = (float*)(Ps + 128 * 128); // [128][128] f32  (64 KB)
  float* mrow = Sf + 128 * 128;
  float* lrow = mrow + 128;
  float* frow = lrow + 128;

  const int t   = threadIdx.x;
  const int qt  = blockIdx.x;
  const int bh  = blockIdx.y;
  const int b   = bh / HEADS;
  const int h   = bh % HEADS;
  const int q0  = qt * 128;
  const size_t base = ((size_t)b * SEQ) * HIDDEN + (size_t)h * HEAD_DIM;

  const int wv   = t >> 5;
  const int lane = t & 31;
  const int cN   = lane & 15;
  const int rb   = (lane >> 4) * 8;

  // --- stage Q block once (async) ------------------------------------------
#pragma unroll
  for (int c = 0; c < 8; ++c) {
    int idx = t + c * 256;
    int r = idx >> 4, cg = (idx & 15) * 8;
    async_copy_b128(&Qs[r * 128 + cg],
                    &Q[base + (size_t)(q0 + r) * HIDDEN + cg]);
  }
  if (t < 128) { mrow[t] = -3.0e38f; lrow[t] = 0.0f; }

  v8f o[8];
#pragma unroll
  for (int j = 0; j < 8; ++j) o[j] = vzero8();

  const float scale = 0.08838834764831845f;   // 1/sqrt(128)

  for (int tk = 0; tk <= qt; ++tk) {
    const int kv0 = tk * 128;
    __syncthreads();   // previous consumers done before restaging

    // --- stage K tile (async; row-major == transposed-B for QK^T) ----------
#pragma unroll
    for (int c = 0; c < 8; ++c) {
      int idx = t + c * 256;
      int kvr = idx >> 4, cg = (idx & 15) * 8;
      async_copy_b128(&Ks[kvr * 128 + cg],
                      &Kg[base + (size_t)(kv0 + kvr) * HIDDEN + cg]);
    }
    // --- stage V transposed: batch loads, then scatter ----------------------
    uint4 vtmp[8];
#pragma unroll
    for (int c = 0; c < 8; ++c) {
      int idx = t + c * 256;
      int kvr = idx >> 4, cg = (idx & 15) * 8;
      vtmp[c] = *(const uint4*)&V[base + (size_t)(kv0 + kvr) * HIDDEN + cg];
    }
#pragma unroll
    for (int c = 0; c < 8; ++c) {
      int idx = t + c * 256;
      int kvr = idx >> 4, cg = (idx & 15) * 8;
      const u16* e = (const u16*)&vtmp[c];
#pragma unroll
      for (int j = 0; j < 8; ++j) Vt[(cg + j) * 128 + kvr] = e[j];
    }
    ASYNC_WAIT();
    __syncthreads();

    // --- S = Q * K^T --------------------------------------------------------
    v8f s[8];
#pragma unroll
    for (int j = 0; j < 8; ++j) s[j] = vzero8();
#pragma unroll
    for (int ks = 0; ks < 128; ks += 32) {
      v16bf a = load_a_frag(Qs, wv * 16, 128, ks);
#pragma unroll
      for (int j = 0; j < 8; ++j) {
        v16bf bj = load_b_frag(Ks, j * 16, 128, ks);
        s[j] = wmma_bf16(a, bj, s[j]);
      }
    }
#pragma unroll
    for (int j = 0; j < 8; ++j)
#pragma unroll
      for (int r = 0; r < 8; ++r) {
        int row = wv * 16 + rb + r;
        int col = j * 16 + cN;
        float v = s[j][r] * scale;
        if (tk == qt && col > row) v = -1.0e9f;   // causal mask
        Sf[row * 128 + col] = v;
      }
    __syncthreads();

    // --- online softmax: one thread per q row -------------------------------
    if (t < 128) {
      const int row = t;
      float mo = mrow[row];
      float mx = mo;
      for (int c2 = 0; c2 < 128; ++c2) mx = fmaxf(mx, Sf[row * 128 + c2]);
      float f = __expf(mo - mx);
      float sum = 0.0f;
      for (int c2 = 0; c2 < 128; ++c2) {
        float p = __expf(Sf[row * 128 + c2] - mx);
        sum += p;
        Ps[row * 128 + c2] = f2bf(p);
      }
      lrow[row] = lrow[row] * f + sum;
      mrow[row] = mx;
      frow[row] = f;
    }
    __syncthreads();

    // --- rescale O, then O += P * V -----------------------------------------
    float fr[8];
#pragma unroll
    for (int r = 0; r < 8; ++r) fr[r] = frow[wv * 16 + rb + r];
#pragma unroll
    for (int j = 0; j < 8; ++j)
#pragma unroll
      for (int r = 0; r < 8; ++r) o[j][r] *= fr[r];

#pragma unroll
    for (int ks = 0; ks < 128; ks += 32) {
      v16bf a = load_a_frag(Ps, wv * 16, 128, ks);
#pragma unroll
      for (int j = 0; j < 8; ++j) {
        v16bf bj = load_b_frag(Vt, j * 16, 128, ks);
        o[j] = wmma_bf16(a, bj, o[j]);
      }
    }
  }

  // --- normalize and store --------------------------------------------------
  float il[8];
#pragma unroll
  for (int r = 0; r < 8; ++r) il[r] = 1.0f / lrow[wv * 16 + rb + r];
#pragma unroll
  for (int j = 0; j < 8; ++j)
#pragma unroll
    for (int r = 0; r < 8; ++r) {
      int row = wv * 16 + rb + r;
      int col = j * 16 + cN;
      Og[base + (size_t)(q0 + row) * HIDDEN + col] = f2bf(o[j][r] * il[r]);
    }
}

// ---------------------------------------------------------------------------
// Launcher
// ---------------------------------------------------------------------------
extern "C" void kernel_launch(void* const* d_in, const int* in_sizes, int n_in,
                              void* d_out, int out_size, void* d_ws,
                              size_t ws_size, hipStream_t stream) {
  const float* x  = (const float*)d_in[0];
  const float* Wq = (const float*)d_in[1];
  const float* Wk = (const float*)d_in[2];
  const float* Wv = (const float*)d_in[3];
  const float* Wo = (const float*)d_in[4];
  float* out = (float*)d_out;

  char* ws = (char*)d_ws;
  const size_t SX = (size_t)MROWS * HIDDEN * sizeof(u16);
  const size_t SW = (size_t)HIDDEN * HIDDEN * sizeof(u16);
  u16* xb  = (u16*)(ws);
  u16* wqb = (u16*)(ws + SX);
  u16* wkb = (u16*)(ws + SX + 1 * SW);
  u16* wvb = (u16*)(ws + SX + 2 * SW);
  u16* wob = (u16*)(ws + SX + 3 * SW);
  u16* Qb  = (u16*)(ws + SX + 4 * SW);
  u16* Kb  = (u16*)(ws + SX + 4 * SW + 1 * SX);
  u16* Vb  = (u16*)(ws + SX + 4 * SW + 2 * SX);
  u16* Ab  = (u16*)(ws + SX + 4 * SW + 3 * SX);

  cvt_f32_bf16_kernel<<<4096, 256, 0, stream>>>(x, xb, (size_t)MROWS * HIDDEN);
  cvt_f32_bf16_kernel<<<2048, 256, 0, stream>>>(Wq, wqb, (size_t)HIDDEN * HIDDEN);
  cvt_f32_bf16_kernel<<<2048, 256, 0, stream>>>(Wk, wkb, (size_t)HIDDEN * HIDDEN);
  cvt_f32_bf16_kernel<<<2048, 256, 0, stream>>>(Wv, wvb, (size_t)HIDDEN * HIDDEN);
  cvt_f32_bf16_kernel<<<2048, 256, 0, stream>>>(Wo, wob, (size_t)HIDDEN * HIDDEN);

  dim3 gg(HIDDEN / 128, MROWS / 256);
  gemm_bf16_kernel<<<gg, 256, 0, stream>>>(xb, wqb, Qb, nullptr,
                                           MROWS, HIDDEN, HIDDEN);
  gemm_bf16_kernel<<<gg, 256, 0, stream>>>(xb, wkb, Kb, nullptr,
                                           MROWS, HIDDEN, HIDDEN);
  gemm_bf16_kernel<<<gg, 256, 0, stream>>>(xb, wvb, Vb, nullptr,
                                           MROWS, HIDDEN, HIDDEN);

  rope_kernel<<<(BATCH * SEQ * HEADS * 64) / 256, 256, 0, stream>>>(Qb, Kb);

  dim3 ga(SEQ / 128, BATCH * HEADS);
  size_t shmem = 4 * (128 * 128 * sizeof(u16)) + 128 * 128 * sizeof(float)
               + 3 * 128 * sizeof(float);
  attn_kernel<<<ga, 256, shmem, stream>>>(Qb, Kb, Vb, Ab);

  gemm_bf16_kernel<<<gg, 256, 0, stream>>>(Ab, wob, nullptr, out,
                                           MROWS, HIDDEN, HIDDEN);
}
